// JointLoss_5815385719356
// MI455X (gfx1250) — compile-verified, matrix operand order checked by reading, not run on hardware
//
#include <hip/hip_runtime.h>

#define NROWS  4096
#define DIM    512
#define HALF_B 2048
#define INV_TAU 10.0f

typedef __attribute__((ext_vector_type(16))) __bf16 v16bf;
typedef __attribute__((ext_vector_type(4)))  __bf16 v4bf;
typedef __attribute__((ext_vector_type(8)))  float  v8f;

union Frag {
  v16bf v;
  uint4 q[2];
};

// ---------------------------------------------------------------------------
// Kernel 0: one-shot split-bf16 decomposition of R: x = hi + lo.
// 8 MB read + 8 MB written; results stay hot in the 192 MB L2 for the GEMM.
// ---------------------------------------------------------------------------
__global__ __launch_bounds__(256)
void convert_kernel(const float* __restrict__ R,
                    __bf16* __restrict__ Rhi, __bf16* __restrict__ Rlo) {
  int i = (blockIdx.x * 256 + threadIdx.x) * 4;
  float4 x = *(const float4*)(R + i);
  float f[4] = {x.x, x.y, x.z, x.w};
  v4bf h, l;
  #pragma unroll
  for (int e = 0; e < 4; ++e) {
    __bf16 hh = (__bf16)f[e];
    h[e] = hh;
    l[e] = (__bf16)(f[e] - (float)hh);
  }
  *(v4bf*)(Rhi + i) = h;
  *(v4bf*)(Rlo + i) = l;
}

// ---------------------------------------------------------------------------
// Kernel 1: fused sim = R R^T tile GEMM + online logsumexp + positive capture.
// PRE=true : B fragments load pre-split bf16 (no VALU in inner loop).
// PRE=false: fallback, converts fp32->bf16 hi/lo on the fly (ws too small).
// Three independent WMMA accumulator chains (hh / hl / lh) so the matrix pipe
// never stalls on an accumulate RAW between consecutive WMMAs.
// ---------------------------------------------------------------------------
template <bool PRE>
__global__ __launch_bounds__(256)
void simclr_loss_kernel(const float* __restrict__ R,
                        const __bf16* __restrict__ Rhi,
                        const __bf16* __restrict__ Rlo,
                        float* __restrict__ partial) {
  __shared__ __bf16 lds_ahi[16][DIM];   // 16 KB
  __shared__ __bf16 lds_alo[16][DIM];   // 16 KB
  __shared__ float  wgm[8][16];
  __shared__ float  wgs[8][16];
  __shared__ float  lds_pos[16];
  __shared__ float  lds_loss[16];

  const int tid  = threadIdx.x;
  const int lane = tid & 31;
  const int wave = tid >> 5;
  const int r0   = blockIdx.x * 16;

  // Stage this block's 16-row A stripe into LDS as split bf16.
  if (PRE) {
    for (int c = tid; c < 16 * DIM / 8; c += 256) {
      int r = c >> 6;              // DIM/8 == 64 chunks per row
      int k = (c & 63) << 3;
      *(uint4*)&lds_ahi[r][k] = *(const uint4*)&Rhi[(size_t)(r0 + r) * DIM + k];
      *(uint4*)&lds_alo[r][k] = *(const uint4*)&Rlo[(size_t)(r0 + r) * DIM + k];
    }
  } else {
    for (int idx = tid; idx < 16 * DIM; idx += 256) {
      int r = idx >> 9;
      int k = idx & (DIM - 1);
      float x = R[(size_t)(r0 + r) * DIM + k];
      __bf16 h = (__bf16)x;
      lds_ahi[r][k] = h;
      lds_alo[r][k] = (__bf16)(x - (float)h);
    }
  }
  __syncthreads();

  // Fragment geometry (16x16x32 bf16 WMMA, wave32):
  //   lanes 0-15  : row/col = lane,    K-halves {0..7, 16..23}
  //   lanes 16-31 : row/col = lane-16, K-halves {8..15, 24..31}
  const int mr    = lane & 15;
  const int half  = lane >> 4;
  const int kbase = half * 8;

  float rm[8], rs[8];
  #pragma unroll
  for (int v = 0; v < 8; ++v) { rm[v] = -3.0e38f; rs[v] = 0.0f; }

  for (int ct = wave; ct < NROWS / 16; ct += 8) {
    const float*  rowB = R   + (size_t)(ct * 16 + mr) * DIM;
    const __bf16* bh   = Rhi + (size_t)(ct * 16 + mr) * DIM;
    const __bf16* bl   = Rlo + (size_t)(ct * 16 + mr) * DIM;
    v8f acc_hh = {};
    v8f acc_hl = {};
    v8f acc_lh = {};
    #pragma unroll 4
    for (int kk = 0; kk < DIM; kk += 32) {
      Frag ahi, alo, bhi, blo;
      ahi.q[0] = *(const uint4*)&lds_ahi[mr][kk + kbase];
      ahi.q[1] = *(const uint4*)&lds_ahi[mr][kk + kbase + 16];
      alo.q[0] = *(const uint4*)&lds_alo[mr][kk + kbase];
      alo.q[1] = *(const uint4*)&lds_alo[mr][kk + kbase + 16];

      if (PRE) {
        bhi.q[0] = *(const uint4*)(bh + kk + kbase);
        bhi.q[1] = *(const uint4*)(bh + kk + kbase + 16);
        blo.q[0] = *(const uint4*)(bl + kk + kbase);
        blo.q[1] = *(const uint4*)(bl + kk + kbase + 16);
      } else {
        float4 t0 = *(const float4*)(rowB + kk + kbase);
        float4 t1 = *(const float4*)(rowB + kk + kbase + 4);
        float4 t2 = *(const float4*)(rowB + kk + kbase + 16);
        float4 t3 = *(const float4*)(rowB + kk + kbase + 20);
        float fb[16] = {t0.x, t0.y, t0.z, t0.w, t1.x, t1.y, t1.z, t1.w,
                        t2.x, t2.y, t2.z, t2.w, t3.x, t3.y, t3.z, t3.w};
        #pragma unroll
        for (int e = 0; e < 16; ++e) {
          __bf16 h = (__bf16)fb[e];
          bhi.v[e] = h;
          blo.v[e] = (__bf16)(fb[e] - (float)h);
        }
      }
      // split-bf16 fp32 emulation: hi*hi + hi*lo + lo*hi,
      // each on its own accumulator chain (independent WMMAs).
      acc_hh = __builtin_amdgcn_wmma_f32_16x16x32_bf16(false, ahi.v, false, bhi.v,
                                                       (short)0, acc_hh, false, false);
      acc_hl = __builtin_amdgcn_wmma_f32_16x16x32_bf16(false, ahi.v, false, blo.v,
                                                       (short)0, acc_hl, false, false);
      acc_lh = __builtin_amdgcn_wmma_f32_16x16x32_bf16(false, alo.v, false, bhi.v,
                                                       (short)0, acc_lh, false, false);
    }
    // Combine the three chains (small terms first for rounding).
    v8f acc = (acc_hl + acc_lh) + acc_hh;

    // Online logsumexp update + positive capture.
    // acc element v -> (row = v + 8*half, col = ct*16 + mr)
    #pragma unroll
    for (int v = 0; v < 8; ++v) {
      int mrow = v + half * 8;
      int i = r0 + mrow;
      int j = ct * 16 + mr;
      float x = acc[v] * INV_TAU;
      int posc = (i < HALF_B) ? i + HALF_B : i - HALF_B;
      if (j == posc) lds_pos[mrow] = x;
      if (j != i) {
        float mo = rm[v];
        float mn = fmaxf(mo, x);
        rs[v] = rs[v] * __expf(mo - mn) + __expf(x - mn);
        rm[v] = mn;
      }
    }
  }

  // Butterfly (max,sumexp) merge across each 16-lane group.
  #pragma unroll
  for (int v = 0; v < 8; ++v) {
    float m = rm[v], s = rs[v];
    #pragma unroll
    for (int off = 8; off >= 1; off >>= 1) {
      float m2 = __shfl_xor(m, off, 32);
      float s2 = __shfl_xor(s, off, 32);
      float mn = fmaxf(m, m2);
      s = s * __expf(m - mn) + s2 * __expf(m2 - mn);
      m = mn;
    }
    if (mr == 0) {
      wgm[wave][v + half * 8] = m;
      wgs[wave][v + half * 8] = s;
    }
  }
  __syncthreads();

  if (tid < 16) {
    float m = wgm[0][tid], s = wgs[0][tid];
    #pragma unroll
    for (int w = 1; w < 8; ++w) {
      float m2 = wgm[w][tid], s2 = wgs[w][tid];
      float mn = fmaxf(m, m2);
      s = s * __expf(m - mn) + s2 * __expf(m2 - mn);
      m = mn;
    }
    lds_loss[tid] = m + __logf(s) - lds_pos[tid];
  }
  __syncthreads();

  if (tid == 0) {
    float a = 0.f;
    #pragma unroll
    for (int r = 0; r < 16; ++r) a += lds_loss[r];
    partial[blockIdx.x] = a * (1.0f / (float)NROWS);
  }
}

// Deterministic fixed-order final sum (no float atomics).
__global__ void final_reduce_kernel(const float* __restrict__ partial,
                                    float* __restrict__ out) {
  if (threadIdx.x == 0) {
    float a = 0.f;
    for (int i = 0; i < NROWS / 16; ++i) a += partial[i];
    out[0] = a;
  }
}

extern "C" void kernel_launch(void* const* d_in, const int* in_sizes, int n_in,
                              void* d_out, int out_size, void* d_ws, size_t ws_size,
                              hipStream_t stream) {
  const float* R = (const float*)d_in[0];
  float* out     = (float*)d_out;

  // ws layout: [256 partial floats (1 KB) | Rhi (4 MB) | Rlo (4 MB)]
  const size_t elems    = (size_t)NROWS * DIM;
  const size_t hi_off   = 1024;
  const size_t lo_off   = hi_off + elems * sizeof(__bf16);
  const size_t ws_need  = lo_off + elems * sizeof(__bf16);

  float*  part = (float*)d_ws;
  __bf16* rhi  = (__bf16*)((char*)d_ws + hi_off);
  __bf16* rlo  = (__bf16*)((char*)d_ws + lo_off);

  if (ws_size >= ws_need) {
    convert_kernel<<<(int)(elems / (256 * 4)), 256, 0, stream>>>(R, rhi, rlo);
    simclr_loss_kernel<true><<<NROWS / 16, 256, 0, stream>>>(R, rhi, rlo, part);
  } else {
    simclr_loss_kernel<false><<<NROWS / 16, 256, 0, stream>>>(R, rhi, rlo, part);
  }
  final_reduce_kernel<<<1, 32, 0, stream>>>(part, out);
}
